// MySubClassModel_47382079209761
// MI455X (gfx1250) — matrix operand order repaired and checked by reading, not compiled, not run on hardware
//
#include <hip/hip_runtime.h>
#include <hip/hip_bf16.h>

typedef __attribute__((ext_vector_type(16))) _Float16 v16h;
typedef __attribute__((ext_vector_type(8)))  _Float16 v8h;
typedef __attribute__((ext_vector_type(8)))  float    v8f;
typedef __attribute__((ext_vector_type(4)))  unsigned int u32x4;
typedef int i32x4v __attribute__((vector_size(16)));   // matches builtin param type

#define NB   64
#define NN   14
#define KCH  512
#define NCLS 10
#define POS  196            // 14*14
#define MROWS (NB*POS)      // 12544
#define KDIM (9*KCH)        // 4608
#define NCHUNK (KDIM/32)    // 144

#define BSTR 40             // B LDS row stride in halves (80B: 16B-aligned, bank-spread)

#if __has_builtin(__builtin_amdgcn_global_load_async_to_lds_b128) && \
    __has_builtin(__builtin_amdgcn_s_wait_asynccnt)
#define USE_ASYNC 1
#endif

__device__ __forceinline__ void copy16(const _Float16* src, _Float16* dst) {
#ifdef USE_ASYNC
    __builtin_amdgcn_global_load_async_to_lds_b128(
        (i32x4v*)src, (i32x4v*)dst, 0, 0);
#else
    *(u32x4*)dst = *(const u32x4*)src;
#endif
}

// ---------------------------------------------------------------------------
// Per-(b,c) spatial argmax, template gather, relu mask. 1 block = 1 batch,
// thread = channel. Strict > keeps first occurrence (matches jnp.argmax).
// ---------------------------------------------------------------------------
__global__ __launch_bounds__(512) void mask_kernel(const float* __restrict__ x,
                                                   const float* __restrict__ tp,
                                                   float* __restrict__ out) {
    int b = blockIdx.x;
    int c = threadIdx.x;
    const float* xb = x + (size_t)b * POS * KCH + c;
    float m = -3.402823466e38f;
    int idx = 0;
    for (int p = 0; p < POS; ++p) {
        float v = xb[(size_t)p * KCH];
        if (v > m) { m = v; idx = p; }
    }
    const float* t = tp + (size_t)c * POS * POS + (size_t)idx * POS;
    float* ob = out + (size_t)b * POS * KCH + c;
    for (int p = 0; p < POS; ++p) {
        float v = xb[(size_t)p * KCH] * t[p];
        ob[(size_t)p * KCH] = v > 0.f ? v : 0.f;
    }
}

// ---------------------------------------------------------------------------
// Zero-padded f16 activations: xpad[b][16][16][512]
// ---------------------------------------------------------------------------
__global__ __launch_bounds__(256) void pad_kernel(const float* __restrict__ x1,
                                                  _Float16* __restrict__ xpad) {
    int i = blockIdx.x * 256 + threadIdx.x;
    int c  = i & 511;
    int r  = i >> 9;
    int xx = r & 15; r >>= 4;
    int yy = r & 15;
    int b  = r >> 4;
    float v = 0.f;
    if (yy >= 1 && yy <= NN && xx >= 1 && xx <= NN)
        v = x1[(((size_t)b * NN + (yy - 1)) * NN + (xx - 1)) * KCH + c];
    xpad[i] = (_Float16)v;
}

// ---------------------------------------------------------------------------
// Weights f32 -> f16 pre-transposed into WMMA B-fragment order:
//   whT[(kc*512 + n)*32 + kin] = w[(kc*32 + kin)*512 + n]
// so each K-chunk's 32KB B tile is one contiguous global block, and each
// lane's fragment is a contiguous 32B run in LDS.
// ---------------------------------------------------------------------------
__global__ __launch_bounds__(256) void convw_t_kernel(const float* __restrict__ w,
                                                      _Float16* __restrict__ whT) {
    int o = blockIdx.x * 256 + threadIdx.x;    // 0 .. 9*512*512-1
    int kin = o & 31;
    int n   = (o >> 5) & 511;
    int kc  = o >> 14;
    whT[o] = (_Float16)w[(size_t)(kc * 32 + kin) * KCH + n];
}

// ---------------------------------------------------------------------------
// Implicit-GEMM 3x3 SAME conv via v_wmma_f32_16x16x32_f16.
// Block = 8 waves: 16 M-rows x 512 oc. Double-buffered LDS staged with
// async global->LDS B128 copies (9 per wave per chunk: 8 for the 32KB B tile,
// 1 for the 1KB A tile), ASYNCcnt-pipelined against the WMMA stream.
// ---------------------------------------------------------------------------
__global__ __launch_bounds__(256) void conv_wmma(const _Float16* __restrict__ xpad,
                                                 const _Float16* __restrict__ whT,
                                                 const float* __restrict__ bias,
                                                 float* __restrict__ h) {
    __shared__ _Float16 As[2][16 * 32];        // 2 x 1KB
    __shared__ _Float16 Bs[2][512 * BSTR];     // 2 x 40KB

    int t    = threadIdx.x;
    int wg   = blockIdx.x;                     // M tile 0..783
    int lane = t & 31;
    int wave = t >> 5;
    int lg   = lane >> 4;
    int lmod = lane & 15;

    // A staging: one B128 per thread (threads >=64 duplicate transfer 63)
    int aa   = t < 64 ? t : 63;
    int arow = aa >> 2;                        // 0..15
    int aseg = aa & 3;                         // 16B segment within 64B row
    int g   = wg * 16 + arow;
    int bb  = g / POS;
    int pos = g % POS;
    int oy  = pos / NN;
    int ox  = pos % NN;

    auto issue = [&](int kc, int buf) {
        int koff = kc * 32;
        int rr   = koff >> 9;                  // tap 0..8
        int ic0  = koff & 511;
        int ky   = rr / 3, kx = rr % 3;
        const _Float16* bsrc = whT + (size_t)kc * (512 * 32);   // contiguous 32KB
        _Float16* bdst = &Bs[buf][0];
        #pragma unroll
        for (int i = 0; i < 8; ++i) {
            int id  = i * 256 + t;             // 0..2047
            int n   = id >> 2;
            int seg = id & 3;
            copy16(bsrc + id * 8, bdst + n * BSTR + seg * 8);
        }
        const _Float16* asrc =
            xpad + ((size_t)(bb * 16 + oy + ky) * 16 + (ox + kx)) * KCH + ic0 + aseg * 8;
        copy16(asrc, &As[buf][arow * 32 + aseg * 8]);
        if (kc + 1 < NCHUNK)                   // warm L2 for the chunk after
            __builtin_prefetch(bsrc + 512 * 32 + t * 8, 0, 2);
    };

    v8f acc[4];
    #pragma unroll
    for (int i = 0; i < 4; ++i) acc[i] = v8f{0.f,0.f,0.f,0.f,0.f,0.f,0.f,0.f};

    issue(0, 0);
    for (int kc = 0; kc < NCHUNK; ++kc) {
        int cur = kc & 1;
        if (kc + 1 < NCHUNK) {
            issue(kc + 1, cur ^ 1);
#ifdef USE_ASYNC
            __builtin_amdgcn_s_wait_asynccnt(9);   // chunk kc's 9 copies done
#endif
        } else {
#ifdef USE_ASYNC
            __builtin_amdgcn_s_wait_asynccnt(0);
#endif
        }
        __syncthreads();                       // all waves' staging visible

        // A fragment (contiguous K order, matches B): lane row m=lmod;
        // lg=0 -> K 0..15, lg=1 -> K 16..31
        const _Float16* ab = &As[cur][0];
        v8h alo = *(const v8h*)(ab + lmod * 32 + lg * 16);
        v8h ahi = *(const v8h*)(ab + lmod * 32 + lg * 16 + 8);
        v16h a = __builtin_shufflevector(alo, ahi,
                                         0,1,2,3,4,5,6,7,8,9,10,11,12,13,14,15);

        int nb0 = wave * 64;
        #pragma unroll
        for (int ni = 0; ni < 4; ++ni) {
            int n = nb0 + ni * 16 + lmod;      // B column for this lane
            const _Float16* bp = &Bs[cur][n * BSTR + lg * 16];
            v8h blo = *(const v8h*)(bp);
            v8h bhi = *(const v8h*)(bp + 8);
            v16h bf = __builtin_shufflevector(blo, bhi,
                                              0,1,2,3,4,5,6,7,8,9,10,11,12,13,14,15);
            acc[ni] = __builtin_amdgcn_wmma_f32_16x16x32_f16(
                false, a, false, bf, (short)0, acc[ni], false, false);
        }
        __syncthreads();                       // done reading buf before reuse
    }

    // epilogue: bias + relu. C/D layout: lane -> N=lmod, VGPR r -> M=r+8*lg
    int nbase = wave * 64;
    #pragma unroll
    for (int ni = 0; ni < 4; ++ni) {
        int n = nbase + ni * 16 + lmod;
        float bv = bias[n];
        #pragma unroll
        for (int r = 0; r < 8; ++r) {
            int grow = wg * 16 + r + 8 * lg;
            float v = acc[ni][r] + bv;
            h[(size_t)grow * KCH + n] = v > 0.f ? v : 0.f;
        }
    }
}

// ---------------------------------------------------------------------------
// Fused 2x2 maxpool + dense(25088x10) + softmax. 1 block per batch.
// ---------------------------------------------------------------------------
__global__ __launch_bounds__(256) void pool_dense_softmax(const float* __restrict__ x2,
                                                          const float* __restrict__ dw,
                                                          const float* __restrict__ db,
                                                          float* __restrict__ out) {
    __shared__ float s[256 * NCLS];
    int b = blockIdx.x, t = threadIdx.x;
    float acc[NCLS];
    #pragma unroll
    for (int j = 0; j < NCLS; ++j) acc[j] = 0.f;

    for (int f = t; f < 7 * 7 * KCH; f += 256) {
        int c  = f & 511;
        int pp = f >> 9;
        int py = pp / 7, px = pp % 7;
        size_t base = (((size_t)b * NN + 2 * py) * NN + 2 * px) * KCH + c;
        float v0 = x2[base], v1 = x2[base + KCH];
        float v2 = x2[base + NN * KCH], v3 = x2[base + NN * KCH + KCH];
        float m = fmaxf(fmaxf(v0, v1), fmaxf(v2, v3));
        const float* w = dw + (size_t)f * NCLS;
        #pragma unroll
        for (int j = 0; j < NCLS; ++j) acc[j] += m * w[j];
    }
    #pragma unroll
    for (int j = 0; j < NCLS; ++j) s[t * NCLS + j] = acc[j];
    __syncthreads();
    for (int stp = 128; stp > 0; stp >>= 1) {
        if (t < stp) {
            #pragma unroll
            for (int j = 0; j < NCLS; ++j) s[t * NCLS + j] += s[(t + stp) * NCLS + j];
        }
        __syncthreads();
    }
    if (t == 0) {
        float l[NCLS];
        float mx = -3.402823466e38f;
        #pragma unroll
        for (int j = 0; j < NCLS; ++j) { l[j] = s[j] + db[j]; mx = fmaxf(mx, l[j]); }
        float sum = 0.f;
        #pragma unroll
        for (int j = 0; j < NCLS; ++j) { l[j] = __expf(l[j] - mx); sum += l[j]; }
        #pragma unroll
        for (int j = 0; j < NCLS; ++j) out[b * NCLS + j] = l[j] / sum;
    }
}

// ---------------------------------------------------------------------------
extern "C" void kernel_launch(void* const* d_in, const int* in_sizes, int n_in,
                              void* d_out, int out_size, void* d_ws, size_t ws_size,
                              hipStream_t stream) {
    (void)in_sizes; (void)n_in; (void)out_size; (void)ws_size;
    const float* x  = (const float*)d_in[0];   // (64,14,14,512)
    const float* tp = (const float*)d_in[1];   // (512,14,14,14,14)
    const float* cw = (const float*)d_in[2];   // (3,3,512,512)
    const float* cb = (const float*)d_in[3];   // (512,)
    const float* dw = (const float*)d_in[4];   // (25088,10)
    const float* db = (const float*)d_in[5];   // (10,)

    float* out = (float*)d_out;                // 640 softmax values
    float* x1  = out + NB * NCLS;
    float* x2  = x1 + (size_t)NB * POS * KCH;

    char* ws = (char*)d_ws;
    _Float16* xpad = (_Float16*)ws;                        // 16 MB
    _Float16* whT  = (_Float16*)(ws + 16777216);           // 4.5 MB (fragment order)
    float*    h    = (float*)(ws + 21495808);              // 25.7 MB

    mask_kernel<<<NB, 512, 0, stream>>>(x, tp, x1);
    pad_kernel<<<(NB * 16 * 16 * KCH) / 256, 256, 0, stream>>>(x1, xpad);
    convw_t_kernel<<<(9 * KCH * KCH) / 256, 256, 0, stream>>>(cw, whT);
    conv_wmma<<<MROWS / 16, 256, 0, stream>>>(xpad, whT, cb, h);
    mask_kernel<<<NB, 512, 0, stream>>>(h, tp, x2);
    pool_dense_softmax<<<NB, 256, 0, stream>>>(x2, dw, db, out);
}
